// CircuitSAT_74225624809872
// MI455X (gfx1250) — matrix operand order repaired and checked by reading, not compiled, not run on hardware
//
#include <hip/hip_runtime.h>
#include <hip/hip_bf16.h>

typedef __bf16 bf16;
typedef __attribute__((ext_vector_type(16))) __bf16 v16bf;
typedef __attribute__((ext_vector_type(8)))  float  v8f;

#define DP   128   // D=100 padded
#define DAP  64    // DA=50 padded
#define GP   320   // 3*D=300 padded (multiple of 64 for 4-tile wave strips)

// ---------------- WMMA A fragment loader (ISA 7.12.2 layout) ----------------
// A: 16x32 bf16, row-major source. lane l: row = l&15; lanes 0-15 hold
// K {0..7,16..23}, lanes 16-31 hold K {8..15,24..31}. Two 16B contiguous loads.
__device__ __forceinline__ v16bf load_a_frag(const bf16* A, int lda, int k0, int lane) {
    const bf16* p = A + (size_t)(lane & 15) * lda + k0 + ((lane >> 4) << 3);
    v16bf a;
#pragma unroll
    for (int i = 0; i < 8; ++i) a[i] = p[i];
#pragma unroll
    for (int i = 0; i < 8; ++i) a[8 + i] = p[16 + i];
    return a;
}

// ---------------- generic WMMA GEMM: C = act(A @ Bpacked + bias) -------------
// Bp fragment-packed: tile (kt,nt), lane l's 16 bf16 values contiguous at
// ((nt*KT + kt)*32 + l)*16. Each wave computes a 16x64 C strip (4 n-tiles).
__global__ __launch_bounds__(256) void wmma_gemm(
    const bf16* __restrict__ A, int lda, const bf16* __restrict__ Bp,
    const float* __restrict__ bias, void* __restrict__ C, int ldc,
    int M, int Nn, int K, int act, int out_bf16)
{
    const int KT      = K >> 5;
    const int ngroups = Nn >> 6;
    const int wave    = blockIdx.x * 8 + (threadIdx.x >> 5);
    const int tm = wave / ngroups;
    const int ng = wave - tm * ngroups;
    if (tm * 16 >= M) return;            // wave-uniform: EXEC stays all-ones
    const int lane = threadIdx.x & 31;

    const bf16*  Arow = A + (size_t)(tm * 16) * lda;
    const v16bf* Bf   = (const v16bf*)Bp;
    v8f acc[4] = {{}, {}, {}, {}};
    for (int kt = 0; kt < KT; ++kt) {
        v16bf af = load_a_frag(Arow, lda, kt * 32, lane);
#pragma unroll
        for (int j = 0; j < 4; ++j) {
            v16bf bfr = Bf[(size_t)((ng * 4 + j) * KT + kt) * 32 + lane];
            acc[j] = __builtin_amdgcn_wmma_f32_16x16x32_bf16(
                false, af, false, bfr, (short)0, acc[j], false, false);
        }
    }
    const int rb = tm * 16 + ((lane >> 4) << 3);
#pragma unroll
    for (int j = 0; j < 4; ++j) {
        const int col = (ng * 4 + j) * 16 + (lane & 15);
        const float bv = bias[col];
#pragma unroll
        for (int v = 0; v < 8; ++v) {
            float x = acc[j][v] + bv;
            if (act) x = fmaxf(x, 0.0f);
            if (out_bf16) ((bf16*)C)[(size_t)(rb + v) * ldc + col] = (bf16)x;
            else          ((float*)C)[(size_t)(rb + v) * ldc + col] = x;
        }
    }
}

// ---------------- fused GRU: gi/gh strip GEMMs in LDS + gates ----------------
// One block (4 waves) per 16-row tile. Stages X (f32 msg -> bf16) and H (bf16)
// tiles in LDS, computes gi = X@Wih.T+bih and gh = H@Whh.T+bhh into LDS via
// 10 WMMA strip jobs (5 n-groups x {gi,gh}), then applies GRU gates from LDS.
// gi/gh never touch HBM.
__global__ __launch_bounds__(128) void fused_gru(
    const float* __restrict__ X, const bf16* __restrict__ WihP,
    const bf16* __restrict__ WhhP, const float* __restrict__ bih,
    const float* __restrict__ bhh, float* __restrict__ h, bf16* __restrict__ hb)
{
    __shared__ float sGI[16 * GP];
    __shared__ float sGH[16 * GP];
    __shared__ bf16  sX[16 * DP];
    __shared__ bf16  sH[16 * DP];

    const int tm   = blockIdx.x;
    const int tid  = threadIdx.x;
    const int wv   = tid >> 5;
    const int lane = tid & 31;

    // stage input tiles (contiguous 16x128 panels)
    const float* Xg = X + (size_t)tm * 16 * DP;
    const bf16*  Hg = hb + (size_t)tm * 16 * DP;
    for (int i = tid; i < 16 * DP; i += 128) {
        sX[i] = (bf16)Xg[i];
        sH[i] = Hg[i];
    }
    __syncthreads();

    // 10 strip jobs: job = ng*2 + mat  (mat 0 = gi, 1 = gh)
    for (int job = wv; job < 10; job += 4) {
        const int mat = job & 1;
        const int ng  = job >> 1;
        const bf16*  At   = mat ? sH : sX;
        const v16bf* Bf   = (const v16bf*)(mat ? WhhP : WihP);
        const float* bias = mat ? bhh : bih;
        float*       outS = mat ? sGH : sGI;
        v8f acc[4] = {{}, {}, {}, {}};
#pragma unroll
        for (int kt = 0; kt < 4; ++kt) {   // K = DP = 128
            v16bf af = load_a_frag(At, DP, kt * 32, lane);
#pragma unroll
            for (int j = 0; j < 4; ++j) {
                v16bf bfr = Bf[(size_t)((ng * 4 + j) * 4 + kt) * 32 + lane];
                acc[j] = __builtin_amdgcn_wmma_f32_16x16x32_bf16(
                    false, af, false, bfr, (short)0, acc[j], false, false);
            }
        }
        const int rb = (lane >> 4) << 3;
#pragma unroll
        for (int j = 0; j < 4; ++j) {
            const int col = (ng * 4 + j) * 16 + (lane & 15);
            const float bv = bias[col];
#pragma unroll
            for (int v = 0; v < 8; ++v)
                outS[(rb + v) * GP + col] = acc[j][v] + bv;
        }
    }
    __syncthreads();

    // GRU gates: 16 rows x 100 cols out of LDS
    for (int i = tid; i < 16 * 100; i += 128) {
        const int row = i / 100, d = i - row * 100;
        const float ir  = sGI[row * GP + d],       hr = sGH[row * GP + d];
        const float iz  = sGI[row * GP + 100 + d], hz = sGH[row * GP + 100 + d];
        const float in_ = sGI[row * GP + 200 + d], hn = sGH[row * GP + 200 + d];
        const size_t g = (size_t)(tm * 16 + row) * DP + d;
        const float hold = h[g];
        const float r  = 1.0f / (1.0f + __expf(-(ir + hr)));
        const float z  = 1.0f / (1.0f + __expf(-(iz + hz)));
        const float nw = tanhf(in_ + r * hn);
        const float hnew = (1.0f - z) * nw + z * hold;
        h[g]  = hnew;
        hb[g] = (bf16)hnew;
    }
}

// ---- weight packing: f32 -> padded bf16 in WMMA-fragment-ready order --------
__global__ void pack_weight_frag(const float* __restrict__ W, int in_rows, int in_cols,
                                 bf16* __restrict__ out, int K, int Nn, int do_t)
{
    int idx = blockIdx.x * blockDim.x + threadIdx.x;
    if (idx >= K * Nn) return;
    int i    = idx & 15;
    int lane = (idx >> 4) & 31;
    int t    = idx >> 9;
    int KT   = K >> 5;
    int kt = t % KT, nt = t / KT;
    int c = nt * 16 + (lane & 15);
    int k = kt * 32 + ((lane >> 4) << 3) + (i & 7) + ((i >> 3) << 4);
    float v = 0.0f;
    if (do_t) { if (c < in_rows && k < in_cols) v = W[(size_t)c * in_cols + k]; }
    else      { if (k < in_rows && c < in_cols) v = W[(size_t)k * in_cols + c]; }
    out[idx] = (bf16)v;
}

__global__ void pack_bias(const float* __restrict__ b, int n, float* __restrict__ out, int on)
{
    int i = blockIdx.x * blockDim.x + threadIdx.x;
    if (i < on) out[i] = (i < n) ? b[i] : 0.0f;
}

// ---------------- h0 = features @ init_W + init_b (pad cols zeroed) ----------
__global__ void init_h(const float* __restrict__ feat, const float* __restrict__ W,
                       const float* __restrict__ b, float* __restrict__ h,
                       bf16* __restrict__ hb)
{
    int n = blockIdx.x, d = threadIdx.x;           // blockDim = 128
    float v = 0.0f;
    if (d < 100) {
        v = b[d];
#pragma unroll
        for (int k = 0; k < 4; ++k) v += feat[n * 4 + k] * W[k * 100 + d];
    }
    h[(size_t)n * DP + d]  = v;
    hb[(size_t)n * DP + d] = (bf16)v;
}

__global__ void zero_f32(float* __restrict__ p, long n) {
    long i = (long)blockIdx.x * blockDim.x + threadIdx.x;
    for (; i < n; i += (long)gridDim.x * blockDim.x) p[i] = 0.0f;
}

// ---------------- edge scatter-add: dst[di[e]] += src[si[e]] -----------------
__global__ void scatter_add(const float* __restrict__ src, const int* __restrict__ si,
                            const int* __restrict__ di, float* __restrict__ dst, int E)
{
    int tid = blockIdx.x * blockDim.x + threadIdx.x;
    int e = tid >> 7, d = tid & 127;
    if (e >= E || d >= 100) return;
    atomicAdd(dst + (size_t)di[e] * DP + d, src[(size_t)si[e] * DP + d]);
}

// ---------------- classifier: out = relu(h@W1+b1)@W2+b2 ----------------------
__global__ __launch_bounds__(256) void classifier(
    const float* __restrict__ h, const float* __restrict__ W1, const float* __restrict__ b1,
    const float* __restrict__ W2, const float* __restrict__ b2, float* __restrict__ out, int N)
{
    __shared__ float sW1[100 * 30], sb1[30], sW2[30], sb2v;
    for (int i = threadIdx.x; i < 100 * 30; i += blockDim.x) sW1[i] = W1[i];
    if (threadIdx.x < 30) { sb1[threadIdx.x] = b1[threadIdx.x]; sW2[threadIdx.x] = W2[threadIdx.x]; }
    if (threadIdx.x == 0) sb2v = b2[0];
    __syncthreads();
    int n = blockIdx.x * blockDim.x + threadIdx.x;
    if (n >= N) return;
    float acc[30];
#pragma unroll
    for (int j = 0; j < 30; ++j) acc[j] = sb1[j];
    const float* hr = h + (size_t)n * DP;
    for (int k = 0; k < 100; ++k) {
        float hv = hr[k];
#pragma unroll
        for (int j = 0; j < 30; ++j) acc[j] += hv * sW1[k * 30 + j];
    }
    float o = sb2v;
#pragma unroll
    for (int j = 0; j < 30; ++j) o += fmaxf(acc[j], 0.0f) * sW2[j];
    out[n] = o;
}

// ============================================================================
extern "C" void kernel_launch(void* const* d_in, const int* in_sizes, int n_in,
                              void* d_out, int out_size, void* d_ws, size_t ws_size,
                              hipStream_t stream)
{
    const float* features = (const float*)d_in[0];
    const int*   edge_row = (const int*)d_in[1];
    const int*   edge_col = (const int*)d_in[2];
    const float* init_W   = (const float*)d_in[3];
    const float* init_b   = (const float*)d_in[4];
    const float* msgW1[2]  = {(const float*)d_in[5],  (const float*)d_in[9]};
    const float* msgb1[2]  = {(const float*)d_in[6],  (const float*)d_in[10]};
    const float* msgW2[2]  = {(const float*)d_in[7],  (const float*)d_in[11]};
    const float* msgb2[2]  = {(const float*)d_in[8],  (const float*)d_in[12]};
    const float* gruWih[2] = {(const float*)d_in[13], (const float*)d_in[17]};
    const float* gruWhh[2] = {(const float*)d_in[14], (const float*)d_in[18]};
    const float* grubih[2] = {(const float*)d_in[15], (const float*)d_in[19]};
    const float* grubhh[2] = {(const float*)d_in[16], (const float*)d_in[20]};
    const float* clsW1 = (const float*)d_in[21];
    const float* clsb1 = (const float*)d_in[22];
    const float* clsW2 = (const float*)d_in[23];
    const float* clsb2 = (const float*)d_in[24];

    const int N = in_sizes[0] / 4;
    const int E = in_sizes[1];

    // -------- workspace layout --------
    size_t off = 0;
    auto take = [&](size_t bytes) -> char* {
        char* p = (char*)d_ws + off;
        off += (bytes + 255) & ~(size_t)255;
        return p;
    };
    float* hf  = (float*)take((size_t)N * DP * 4);
    bf16*  hb  = (bf16*) take((size_t)N * DP * 2);
    bf16*  z1  = (bf16*) take((size_t)N * DAP * 2);
    float* fm  = (float*)take((size_t)N * DP * 4);
    float* msg = (float*)take((size_t)N * DP * 4);
    bf16 *W1p[2], *W2p[2], *Wihp[2], *Whhp[2];
    float *b1p[2], *b2p[2], *bihp[2], *bhhp[2];
    for (int s = 0; s < 2; ++s) {
        W1p[s]  = (bf16*) take((size_t)DP * DAP * 2);
        W2p[s]  = (bf16*) take((size_t)DAP * DP * 2);
        Wihp[s] = (bf16*) take((size_t)DP * GP * 2);
        Whhp[s] = (bf16*) take((size_t)DP * GP * 2);
        b1p[s]  = (float*)take(DAP * 4);
        b2p[s]  = (float*)take(DP * 4);
        bihp[s] = (float*)take(GP * 4);
        bhhp[s] = (float*)take(GP * 4);
    }

    auto cdiv = [](long a, long b) { return (int)((a + b - 1) / b); };

    // -------- pack weights into WMMA-fragment-ready bf16 layouts --------
    for (int s = 0; s < 2; ++s) {
        pack_weight_frag<<<cdiv(DP * DAP, 256), 256, 0, stream>>>(msgW1[s], 100, 50,  W1p[s],  DP,  DAP, 0);
        pack_weight_frag<<<cdiv(DAP * DP, 256), 256, 0, stream>>>(msgW2[s], 50, 100,  W2p[s],  DAP, DP,  0);
        pack_weight_frag<<<cdiv(DP * GP,  256), 256, 0, stream>>>(gruWih[s], 300, 100, Wihp[s], DP,  GP,  1);
        pack_weight_frag<<<cdiv(DP * GP,  256), 256, 0, stream>>>(gruWhh[s], 300, 100, Whhp[s], DP,  GP,  1);
        pack_bias<<<1, DAP, 0, stream>>>(msgb1[s], 50,  b1p[s],  DAP);
        pack_bias<<<1, DP,  0, stream>>>(msgb2[s], 100, b2p[s],  DP);
        pack_bias<<<cdiv(GP, 256), 256, 0, stream>>>(grubih[s], 300, bihp[s], GP);
        pack_bias<<<cdiv(GP, 256), 256, 0, stream>>>(grubhh[s], 300, bhhp[s], GP);
    }

    // -------- h0 --------
    init_h<<<N, DP, 0, stream>>>(features, init_W, init_b, hf, hb);

    auto gemm = [&](const bf16* A, int lda, const bf16* Bp, const float* bias,
                    void* C, int ldc, int M, int Nn, int K, int act, int outbf) {
        long waves = (long)(M / 16) * (Nn / 64);   // one 16x64 strip per wave
        wmma_gemm<<<cdiv(waves, 8), 256, 0, stream>>>(A, lda, Bp, bias, C, ldc,
                                                      M, Nn, K, act, outbf);
    };

    const int zgrid = 4096;
    // -------- 20 message-passing rounds, forward then backward half-round ----
    for (int r = 0; r < 20; ++r) {
        for (int s = 0; s < 2; ++s) {      // s=0: forward, s=1: backward
            // message MLP: z1 = relu(h@W1+b1) [bf16]; fm = z1@W2+b2 [f32]
            gemm(hb, DP, W1p[s], b1p[s], z1, DAP, N, DAP, DP, 1, 1);
            gemm(z1, DAP, W2p[s], b2p[s], fm, DP, N, DP, DAP, 0, 0);
            // scatter: msg[dst] += fm[src]
            zero_f32<<<zgrid, 256, 0, stream>>>(msg, (long)N * DP);
            const int* si = s == 0 ? edge_col : edge_row;
            const int* di = s == 0 ? edge_row : edge_col;
            scatter_add<<<cdiv((long)E * 128, 256), 256, 0, stream>>>(fm, si, di, msg, E);
            // fused GRU: gi/gh strips in LDS, gates, h update (no HBM gi/gh)
            fused_gru<<<N / 16, 128, 0, stream>>>(msg, Wihp[s], Whhp[s],
                                                  bihp[s], bhhp[s], hf, hb);
        }
    }

    // -------- classifier --------
    classifier<<<cdiv(N, 256), 256, 0, stream>>>(hf, clsW1, clsb1, clsW2, clsb2,
                                                 (float*)d_out, N);
    (void)n_in; (void)out_size; (void)ws_size;
}
